// NodeMessageBlock_79551384256579
// MI455X (gfx1250) — matrix operand order repaired and checked by reading, not compiled, not run on hardware
//
#include <hip/hip_runtime.h>

typedef __attribute__((ext_vector_type(16))) _Float16 v16h;
typedef __attribute__((ext_vector_type(8)))  float    v8f;

#define WAVES 4            // waves/block for the small GEMM kernels
#define EW 8               // waves/block for the persistent edge kernel
#define EDGE_BLOCKS 640
#define INV_SQRT3f 0.57735026918962576f

// padded LDS row strides (halves); 16B-aligned rows; B reads take one half per row
#define LD_M123 72         // 64 cols + 8 pad   (144 B = 36 dwords -> 2-way conflicts max)
#define LD_M4   264        // 256 cols + 8 pad  (528 B = 132 dwords)

__device__ __forceinline__ v8f wmma16(v16h a, v16h b, v8f c) {
  // D = A(16x32 f16) * B(32x16 f16) + C(16x16 f32)
  return __builtin_amdgcn_wmma_f32_16x16x32_f16(false, a, false, b, (short)0, c,
                                                false, false);
}

// ---- A-tile loader from an f16 buffer (LDS), row-major, row stride ld halves.
// lane<16: row=lane, K = kb+{0..7,16..23}; lane>=16: row=lane-16, K = kb+{8..15,24..31}
__device__ __forceinline__ v16h loadA_h(const _Float16* base, int ld, int lane, int kb) {
  int row = lane & 15;
  int k0  = kb + ((lane >> 4) << 3);
  const _Float16* p = base + row * ld + k0;
  v16h a;
#pragma unroll
  for (int i = 0; i < 8; ++i) { a[i] = p[i]; a[i + 8] = p[i + 16]; }
  return a;
}

// ---- B-tile loader (K x N row-major, row stride ld): column N = nb + lane%16,
// lane<16: K = kb+{0..7,16..23}; lane>=16: K = kb+{8..15,24..31}. Works for global or LDS pointers.
__device__ __forceinline__ v16h loadB(const _Float16* W, int ld, int lane, int kb, int nb) {
  const _Float16* p = W + (size_t)(kb + ((lane >> 4) << 3)) * ld + nb + (lane & 15);
  v16h b;
#pragma unroll
  for (int i = 0; i < 8; ++i) { b[i] = p[(size_t)i * ld]; b[i + 8] = p[(size_t)(i + 16) * ld]; }
  return b;
}

// ---- A-tile loader from global f32 (converting to f16), row length ROWLEN floats,
// element K=k at base[row*ROWLEN + off + k*stride]. Caller guarantees full tiles.
template <int ROWLEN>
__device__ __forceinline__ v16h loadA_gf(const float* base, int row0,
                                         int lane, int kb, int off, int stride) {
  int row = row0 + (lane & 15);
  const float* p = base + (size_t)row * ROWLEN + off;
  int k0 = kb + ((lane >> 4) << 3);
  v16h a;
#pragma unroll
  for (int i = 0; i < 8; ++i) {
    a[i]     = (_Float16)p[(k0 + i) * stride];
    a[i + 8] = (_Float16)p[(k0 + i + 16) * stride];
  }
  return a;
}

// ---- D-tile (16x16 f32 acc) store to global (no guards; full tiles guaranteed)
__device__ __forceinline__ void storeD_g(float* out, int rowlen, int row0,
                                         int lane, v8f c, int nb, int colbase, int colstride) {
  int ncol = (lane & 15) + nb;
  int mb   = (lane >> 4) << 3;
  float* p = out + (size_t)(row0 + mb) * rowlen + colbase + ncol * colstride;
#pragma unroll
  for (int r = 0; r < 8; ++r) p[(size_t)r * rowlen] = c[r];
}

// ---- SiLU + f16 store of a D-tile into an LDS buffer (16 x ld halves)
__device__ __forceinline__ void silu_store(_Float16* dst, int ld, int lane, v8f c, int nb) {
  int n  = (lane & 15) + nb;
  int mb = (lane >> 4) << 3;
#pragma unroll
  for (int r = 0; r < 8; ++r) {
    float x = c[r];
    float y = x * (1.0f / (1.0f + __expf(-x)));
    dst[(mb + r) * ld + n] = (_Float16)y;
  }
}

// ---- cooperative copy of a KxC f16 weight into LDS with padded row stride ldp
__device__ __forceinline__ void copy_w(const _Float16* __restrict__ g, _Float16* s,
                                       int rows, int cols, int ldp, int tid, int nthr) {
  int total = rows * cols;                 // cols is a multiple of 8
  for (int i = tid * 8; i < total; i += nthr * 8) {
    int r = i / cols, c = i % cols;
#pragma unroll
    for (int j = 0; j < 8; ++j) s[r * ldp + c + j] = g[i + j];
  }
}

// =============== kernel 0: zero the message accumulator ===============
__global__ void k_zero(float* p, int n) {
  int i = blockIdx.x * blockDim.x + threadIdx.x;
  if (i < n) p[i] = 0.0f;
}

// ====== kernel 1: convert+scale (+pad, two-column-range scales) a weight to f16 ======
__global__ void k_convw(const float* __restrict__ src, _Float16* __restrict__ dst,
                        int rows, int cols, int prows,
                        float scale_lo, float scale_hi, int col_split) {
  int i = blockIdx.x * blockDim.x + threadIdx.x;
  int total = prows * cols;
  if (i >= total) return;
  int r = i / cols, c = i % cols;
  float sc = (c < col_split) ? scale_lo : scale_hi;
  float v = (r < rows) ? src[r * cols + c] * sc : 0.0f;
  dst[i] = (_Float16)v;
}

// =============== kernel 2: node up-projection (s and v) ===============
// upfeat[n][0..63] = s', upfeat[n][64 + u*3 + m] = v'   (1/sqrt(64) folded into weights)
__global__ __launch_bounds__(WAVES * 32)
void k_upproj(const float* __restrict__ nf, const _Float16* __restrict__ Wu0h,
              const _Float16* __restrict__ Wu1h, float* __restrict__ upfeat, int N) {
  int lane = threadIdx.x & 31;
  int wv   = threadIdx.x >> 5;
  int n0   = (blockIdx.x * WAVES + wv) * 16;
  if (n0 >= N) return;  // N is a multiple of 16: active waves own full tiles

  { // scalar channels: A = node_feats[:, 0:64]
    v16h a0 = loadA_gf<256>(nf, n0, lane, 0,  0, 1);
    v16h a1 = loadA_gf<256>(nf, n0, lane, 32, 0, 1);
#pragma unroll
    for (int nt = 0; nt < 4; ++nt) {
      v8f c = {};
      c = wmma16(a0, loadB(Wu0h, 64, lane, 0,  nt * 16), c);
      c = wmma16(a1, loadB(Wu0h, 64, lane, 32, nt * 16), c);
      storeD_g(upfeat, 256, n0, lane, c, nt * 16, 0, 1);
    }
  }
#pragma unroll
  for (int m = 0; m < 3; ++m) { // vector channels, component m: stride-3 K
    v16h a0 = loadA_gf<256>(nf, n0, lane, 0,  64 + m, 3);
    v16h a1 = loadA_gf<256>(nf, n0, lane, 32, 64 + m, 3);
#pragma unroll
    for (int nt = 0; nt < 4; ++nt) {
      v8f c = {};
      c = wmma16(a0, loadB(Wu1h, 64, lane, 0,  nt * 16), c);
      c = wmma16(a1, loadB(Wu1h, 64, lane, 32, nt * 16), c);
      storeD_g(upfeat, 256, n0, lane, c, nt * 16, 64 + m, 3);
    }
  }
}

// ====== kernel 3: persistent fused edge MLP + tensor-product message + scatter =======
// Weights staged in LDS once per block; 2 x 16-edge tiles per wave per iteration.
// msg[n][0:64]=m_s0, [64:128]=m_s1, [128:320]=m_v0 (u*3+m), [320:512]=m_v1
// 1/sqrt(3) pre-folded into M4h columns 192..255 (the w3 block).
__global__ __launch_bounds__(EW * 32)
void k_edge(const float* __restrict__ edge_feats, const float* __restrict__ edge_attrs,
            const int* __restrict__ sender, const int* __restrict__ receiver,
            const _Float16* __restrict__ M1h, const _Float16* __restrict__ M2h,
            const _Float16* __restrict__ M3h, const _Float16* __restrict__ M4h,
            const float* __restrict__ upfeat, float* __restrict__ msg, int E) {
  extern __shared__ char smem[];
  _Float16* sM1 = (_Float16*)smem;            // 32  x LD_M123
  _Float16* sM2 = sM1 + 32 * LD_M123;         // 64  x LD_M123
  _Float16* sM3 = sM2 + 64 * LD_M123;         // 64  x LD_M123
  _Float16* sM4 = sM3 + 64 * LD_M123;         // 64  x LD_M4
  _Float16* hAa = sM4 + 64 * LD_M4;           // EW x 2 x 16 x 64
  _Float16* hBa = hAa + EW * 2 * 16 * 64;     // EW x 2 x 16 x 64
  int*      sSa = (int*)(hBa + EW * 2 * 16 * 64);  // EW x 32
  int*      sRa = sSa + EW * 32;                   // EW x 32
  float*    sYa = (float*)(sRa + EW * 32);         // EW x 32 x 4

  const int tid  = threadIdx.x;
  const int lane = tid & 31;
  const int wv   = tid >> 5;

  // ---- stage the MLP weights into LDS once per block
  copy_w(M1h, sM1, 32, 64,  LD_M123, tid, EW * 32);
  copy_w(M2h, sM2, 64, 64,  LD_M123, tid, EW * 32);
  copy_w(M3h, sM3, 64, 64,  LD_M123, tid, EW * 32);
  copy_w(M4h, sM4, 64, 256, LD_M4,   tid, EW * 32);
  __syncthreads();

  _Float16* hA = hAa + wv * (2 * 16 * 64);
  _Float16* hB = hBa + wv * (2 * 16 * 64);
  int*      sS = sSa + wv * 32;
  int*      sR = sRa + wv * 32;
  float*    sY = sYa + wv * 32 * 4;

  const int n   = lane & 15;
  const int mb  = (lane >> 4) << 3;
  const int nTiles = (E + 31) / 32;

  for (int tile = blockIdx.x * EW + wv; tile < nTiles; tile += gridDim.x * EW) {
    const int e0 = tile * 32;   // two 16-edge tiles
    bool tv[2];
    tv[0] = (e0 + 16) <= E;     // exactly right for any E that is a multiple of 16
    tv[1] = (e0 + 32) <= E;

    // stage per-edge metadata: each lane owns one of the 32 edges (clamped reads)
    {
      int e  = e0 + lane;
      int em = (e < E) ? e : (E - 1);
      sS[lane] = sender[em];
      sR[lane] = receiver[em];
      const float* ea = edge_attrs + (size_t)em * 4;
      sY[lane * 4 + 0] = ea[0]; sY[lane * 4 + 1] = ea[1];
      sY[lane * 4 + 2] = ea[2]; sY[lane * 4 + 3] = ea[3];
    }

    // ---- layer 1: A = edge_feats 16x8 zero-padded to 16x32 (1/sqrt8 folded into M1h)
    v16h aL1[2];
#pragma unroll
    for (int t = 0; t < 2; ++t) {
#pragma unroll
      for (int i = 0; i < 16; ++i) aL1[t][i] = (_Float16)0.0f;
    }
    if (lane < 16) { // lanes 0..15 own K=0..7 (valid) and K=16..23 (pad)
#pragma unroll
      for (int t = 0; t < 2; ++t) {
        int e  = e0 + t * 16 + lane;
        int em = (e < E) ? e : (E - 1);
        const float* p = edge_feats + (size_t)em * 8;
#pragma unroll
        for (int i = 0; i < 8; ++i) aL1[t][i] = (_Float16)p[i];
      }
    }
#pragma unroll
    for (int nt = 0; nt < 4; ++nt) {
      v16h b = loadB(sM1, LD_M123, lane, 0, nt * 16);
#pragma unroll
      for (int t = 0; t < 2; ++t) {
        v8f c = {};
        c = wmma16(aL1[t], b, c);
        silu_store(hA + t * 16 * 64, 64, lane, c, nt * 16);
      }
    }

    // ---- layers 2,3 (per-wave LDS ping-pong; same-wave DS ops are in-order)
#pragma unroll
    for (int layer = 0; layer < 2; ++layer) {
      const _Float16* W   = (layer == 0) ? sM2 : sM3;
      const _Float16* src = (layer == 0) ? hA : hB;
      _Float16*       dst = (layer == 0) ? hB : hA;
      v16h a0[2], a1[2];
#pragma unroll
      for (int t = 0; t < 2; ++t) {
        a0[t] = loadA_h(src + t * 16 * 64, 64, lane, 0);
        a1[t] = loadA_h(src + t * 16 * 64, 64, lane, 32);
      }
#pragma unroll
      for (int nt = 0; nt < 4; ++nt) {
        v16h b0 = loadB(W, LD_M123, lane, 0,  nt * 16);
        v16h b1 = loadB(W, LD_M123, lane, 32, nt * 16);
#pragma unroll
        for (int t = 0; t < 2; ++t) {
          v8f c = {};
          c = wmma16(a0[t], b0, c);
          c = wmma16(a1[t], b1, c);
          silu_store(dst + t * 16 * 64, 64, lane, c, nt * 16);
        }
      }
    }

    // ---- layer 4 (16x64 @ 64x256) fused with the tensor-product message + scatter
    v16h a0[2], a1[2];
#pragma unroll
    for (int t = 0; t < 2; ++t) {
      a0[t] = loadA_h(hA + t * 16 * 64, 64, lane, 0);
      a1[t] = loadA_h(hA + t * 16 * 64, 64, lane, 32);
    }
#pragma unroll
    for (int tt = 0; tt < 4; ++tt) {
      v8f c[2][4];
#pragma unroll
      for (int blk = 0; blk < 4; ++blk) {  // w0,w1,w2,w3 column blocks of M4
        v16h b0 = loadB(sM4, LD_M4, lane, 0,  tt * 16 + blk * 64);
        v16h b1 = loadB(sM4, LD_M4, lane, 32, tt * 16 + blk * 64);
#pragma unroll
        for (int t = 0; t < 2; ++t) {
          v8f acc = {};
          acc = wmma16(a0[t], b0, acc);
          acc = wmma16(a1[t], b1, acc);
          c[t][blk] = acc;
        }
      }
      const int u = tt * 16 + n; // channel this lane owns for rows mb..mb+7
#pragma unroll
      for (int t = 0; t < 2; ++t) {
        if (!tv[t]) continue;  // wave-uniform
#pragma unroll
        for (int r = 0; r < 8; ++r) {
          int eloc = t * 16 + mb + r;
          int   snd = sS[eloc];
          int   rcv = sR[eloc];
          float ys  = sY[eloc * 4 + 0];
          float yv0 = sY[eloc * 4 + 1], yv1 = sY[eloc * 4 + 2], yv2 = sY[eloc * 4 + 3];
          const float* uf = upfeat + (size_t)snd * 256;
          float xs  = uf[u];
          float xv0 = uf[64 + u * 3 + 0];
          float xv1 = uf[64 + u * 3 + 1];
          float xv2 = uf[64 + u * 3 + 2];
          float w0 = c[t][0][r], w1 = c[t][1][r], w2 = c[t][2][r], w3 = c[t][3][r];
          float* mrow = msg + (size_t)rcv * 512;
          atomicAdd(mrow + u, w0 * xs * ys);                              // m_s0
          float dot = xv0 * yv0 + xv1 * yv1 + xv2 * yv2;
          atomicAdd(mrow + 64 + u, w3 * dot);                             // m_s1 (1/sqrt3 folded)
          float t1 = w1 * xs;                                             // m_v0
          atomicAdd(mrow + 128 + u * 3 + 0, t1 * yv0);
          atomicAdd(mrow + 128 + u * 3 + 1, t1 * yv1);
          atomicAdd(mrow + 128 + u * 3 + 2, t1 * yv2);
          float t2 = w2 * ys;                                             // m_v1
          atomicAdd(mrow + 320 + u * 3 + 0, t2 * xv0);
          atomicAdd(mrow + 320 + u * 3 + 1, t2 * xv1);
          atomicAdd(mrow + 320 + u * 3 + 2, t2 * xv2);
        }
      }
    }
  }
}

// =============== kernel 4: output linear (S and V), norm folded into weights =========
__global__ __launch_bounds__(WAVES * 32)
void k_out(const float* __restrict__ msg, const _Float16* __restrict__ Wl0h,
           const _Float16* __restrict__ Wl1h, float* __restrict__ out, int N) {
  int lane = threadIdx.x & 31;
  int wv   = threadIdx.x >> 5;
  int n0   = (blockIdx.x * WAVES + wv) * 16;
  if (n0 >= N) return;  // N multiple of 16: full tiles

  { // out_s = msg[:, :128] @ Wl0   (K = 128 -> 4 chunks)
    v16h a[4];
#pragma unroll
    for (int kc = 0; kc < 4; ++kc)
      a[kc] = loadA_gf<512>(msg, n0, lane, kc * 32, 0, 1);
#pragma unroll
    for (int nt = 0; nt < 4; ++nt) {
      v8f c = {};
#pragma unroll
      for (int kc = 0; kc < 4; ++kc)
        c = wmma16(a[kc], loadB(Wl0h, 64, lane, kc * 32, nt * 16), c);
      storeD_g(out, 256, n0, lane, c, nt * 16, 0, 1);
    }
  }
#pragma unroll
  for (int m = 0; m < 3; ++m) { // out_v component m: K element at msg[n][128 + u*3 + m]
    v16h a[4];
#pragma unroll
    for (int kc = 0; kc < 4; ++kc)
      a[kc] = loadA_gf<512>(msg, n0, lane, kc * 32, 128 + m, 3);
#pragma unroll
    for (int nt = 0; nt < 4; ++nt) {
      v8f c = {};
#pragma unroll
      for (int kc = 0; kc < 4; ++kc)
        c = wmma16(a[kc], loadB(Wl1h, 64, lane, kc * 32, nt * 16), c);
      storeD_g(out, 256, n0, lane, c, nt * 16, 64 + m, 3);
    }
  }
}

extern "C" void kernel_launch(void* const* d_in, const int* in_sizes, int n_in,
                              void* d_out, int out_size, void* d_ws, size_t ws_size,
                              hipStream_t stream) {
  (void)n_in; (void)out_size; (void)ws_size;
  // inputs (setup_inputs order)
  const float* node_feats = (const float*)d_in[1];
  const float* edge_attrs = (const float*)d_in[2];
  const float* edge_feats = (const float*)d_in[3];
  const int*   edge_index = (const int*)  d_in[4];
  const float* W_up0 = (const float*)d_in[5];
  const float* W_up1 = (const float*)d_in[6];
  const float* M1    = (const float*)d_in[7];
  const float* M2    = (const float*)d_in[8];
  const float* M3    = (const float*)d_in[9];
  const float* M4    = (const float*)d_in[10];
  const float* W_lin0 = (const float*)d_in[11];
  const float* W_lin1 = (const float*)d_in[12];
  // W_skip0/1 (d_in[13],[14]) are dead code in the reference.

  const int N = in_sizes[1] / 256;   // 20000
  const int E = in_sizes[3] / 8;     // 320000
  const int* sender   = edge_index;
  const int* receiver = edge_index + E;

  // ---- workspace carve-up
  float* msg    = (float*)d_ws;                      // N*512 f32
  float* upfeat = msg + (size_t)N * 512;             // N*256 f32
  _Float16* wh  = (_Float16*)(upfeat + (size_t)N * 256);
  _Float16* M1h  = wh;             // 32*64  (padded from 8 rows)
  _Float16* M2h  = M1h  + 32 * 64;
  _Float16* M3h  = M2h  + 64 * 64;
  _Float16* M4h  = M3h  + 64 * 64; // 64*256
  _Float16* Wu0h = M4h  + 64 * 256;
  _Float16* Wu1h = Wu0h + 64 * 64;
  _Float16* Wl0h = Wu1h + 64 * 64; // 128*64
  _Float16* Wl1h = Wl0h + 128 * 64;

  // ---- zero accumulator
  { int n = N * 512; k_zero<<<(n + 255) / 256, 256, 0, stream>>>(msg, n); }

  // ---- convert weights to f16, folding all normalization constants
  const float sM1 = 0.35355339059327373f;          // 1/sqrt(8)
  const float s64 = 0.125f;                        // 1/sqrt(64)
  const float sL  = 0.005524271728019903f;         // (1/sqrt(128)) / 16
  auto conv = [&](const float* s, _Float16* d, int r, int c, int pr,
                  float lo, float hi, int split) {
    int t = pr * c;
    k_convw<<<(t + 255) / 256, 256, 0, stream>>>(s, d, r, c, pr, lo, hi, split);
  };
  conv(M1,    M1h,  8,   64, 32,  sM1, sM1, 64);
  conv(M2,    M2h,  64,  64, 64,  s64, s64, 64);
  conv(M3,    M3h,  64,  64, 64,  s64, s64, 64);
  conv(M4,    M4h,  64, 256, 64,  s64, s64 * INV_SQRT3f, 192); // fold 1/sqrt3 into w3
  conv(W_up0, Wu0h, 64,  64, 64,  s64, s64, 64);
  conv(W_up1, Wu1h, 64,  64, 64,  s64, s64, 64);
  conv(W_lin0, Wl0h, 128, 64, 128, sL, sL, 64);
  conv(W_lin1, Wl1h, 128, 64, 128, sL, sL, 64);

  // ---- node up-projection (WMMA)
  {
    int tiles  = (N + 15) / 16;
    int blocks = (tiles + WAVES - 1) / WAVES;
    k_upproj<<<blocks, WAVES * 32, 0, stream>>>(node_feats, Wu0h, Wu1h, upfeat, N);
  }
  // ---- persistent fused edge MLP + message + scatter (LDS weights, WMMA, f32 atomics)
  {
    size_t smem = (size_t)(32 * LD_M123 + 64 * LD_M123 + 64 * LD_M123 + 64 * LD_M4
                           + EW * 2 * 16 * 64 * 2) * sizeof(_Float16)
                + (size_t)(EW * 32 * 2) * sizeof(int)
                + (size_t)(EW * 32 * 4) * sizeof(float);
    k_edge<<<EDGE_BLOCKS, EW * 32, smem, stream>>>(edge_feats, edge_attrs, sender, receiver,
                                                   M1h, M2h, M3h, M4h, upfeat, msg, E);
  }
  // ---- output linear (WMMA)
  {
    int tiles  = (N + 15) / 16;
    int blocks = (tiles + WAVES - 1) / WAVES;
    k_out<<<blocks, WAVES * 32, 0, stream>>>(msg, Wl0h, Wl1h, (float*)d_out, N);
  }
}
